// H3Block_56581899157851
// MI455X (gfx1250) — matrix operand order repaired and checked by reading, not compiled, not run on hardware
//
#include <hip/hip_runtime.h>
#include <hip/hip_fp16.h>

// ---------------- problem constants ----------------
#define S_LEN 1024
#define D_DIM 512
#define HEADS 8
#define DH    64
#define HID   64
#define CHUNK 64
#define NCHUNK (S_LEN / CHUNK)
#define DT_   0.1f

typedef __attribute__((ext_vector_type(16))) _Float16 v16h;
typedef __attribute__((ext_vector_type(8)))  float    v8f;

// ---------------- WMMA helpers (CDNA5 wave32, 16x16x32 f16 -> f32) ----------
// A fragment: 16x32 (MxK) f16. Per ISA 7.12.2: lanes 0-15 -> M=lane, K halves
// split 0..7 / 8..15 (VGPR0..3) and 16..23 / 24..31 (VGPR4..7) by lane half.
template <typename F>
__device__ inline v16h load_a_frag(F f, int lane) {
  v16h a;
  const int half = lane >> 4, m = lane & 15;
#pragma unroll
  for (int e = 0; e < 16; ++e) {
    const int vg = e >> 1, w = e & 1;
    const int k = (vg < 4) ? (2 * vg + w + 8 * half)
                           : (16 + 2 * (vg - 4) + w + 8 * half);
    a[e] = f(m, k);
  }
  return a;
}
// B fragment: 32x16 (KxN) f16. Rows striped across lanes: lane%16 = N,
// K split like A: 0..7 / 8..15 (VGPR0..3) and 16..23 / 24..31 (VGPR4..7) by lane half.
template <typename F>
__device__ inline v16h load_b_frag(F f, int lane) {
  v16h b;
  const int half = lane >> 4, n = lane & 15;
#pragma unroll
  for (int e = 0; e < 16; ++e) {
    const int vg = e >> 1, w = e & 1;
    const int k = (vg < 4) ? (2 * vg + w + 8 * half)
                           : (16 + 2 * (vg - 4) + w + 8 * half);
    b[e] = f(k, n);
  }
  return b;
}
// C/D: element r of v8f at lane -> (M = r + 8*(lane>>4), N = lane&15)
template <typename F>
__device__ inline void store_c_frag(v8f c, F st, int lane) {
  const int half = lane >> 4, n = lane & 15;
#pragma unroll
  for (int r = 0; r < 8; ++r) st(r + 8 * half, n, c[r]);
}

__device__ inline v8f wmma_f16(v16h a, v16h b, v8f c) {
  return __builtin_amdgcn_wmma_f32_16x16x32_f16(
      /*neg_a=*/false, a, /*neg_b=*/false, b,
      /*c_mod=*/(short)0, c, /*reuse_a=*/false, /*reuse_b=*/false);
}

// ---------------- Tensor Data Mover (gfx1250) -------------------------------
// 2-D tile DMA global->LDS per CDNA5 ISA (D# groups 0/1).
// Toolchain arity differs: therock-10.0 headers imply the 6-arg builtin.
#if __has_builtin(__builtin_amdgcn_tensor_load_to_lds) && \
    __has_builtin(__builtin_amdgcn_s_wait_tensorcnt)
#define HAVE_TDM 1
typedef __attribute__((ext_vector_type(4))) unsigned int u32x4;
typedef __attribute__((ext_vector_type(8))) int i32x8;
typedef __attribute__((ext_vector_type(4))) int i32x4;

__device__ inline void tdm_load_tile_2d(unsigned lds_off, const void* gptr,
                                        unsigned tile_w, unsigned tile_h,
                                        unsigned row_stride_elems) {
  const unsigned long long ga = (unsigned long long)(uintptr_t)gptr;
  u32x4 g0;
  g0[0] = 1u;                                   // count=1 (valid user D#)
  g0[1] = lds_off;                              // LDS byte address
  g0[2] = (unsigned)(ga & 0xffffffffu);         // global_addr lo
  g0[3] = (unsigned)((ga >> 32) & 0x01ffffffu)  // global_addr hi (57b)
          | (2u << 30);                         // type=2 ("image")
  // group1: data_size=2 (4B); tensor dims == tile dims (stride may exceed)
  i32x8 g1;
  g1[0] = (int)(2u << 16);                                    // data_size
  g1[1] = (int)((tile_w & 0xffffu) << 16);                    // tdim0[15:0]
  g1[2] = (int)(((tile_w >> 16) & 0xffffu) |
                ((tile_h & 0xffffu) << 16));                  // tdim0hi|tdim1lo
  g1[3] = (int)(((tile_h >> 16) & 0xffffu) |
                ((tile_w & 0xffffu) << 16));                  // tdim1hi|tile0
  g1[4] = (int)(tile_h & 0xffffu);                            // tile1|tile2=0
  g1[5] = (int)row_stride_elems;                              // dim0_stride lo
  g1[6] = 0;                                                  // hi|dim1_stride
  g1[7] = 0;
  i32x4 gz = {0, 0, 0, 0};
#if __has_include(<hip/amd_detail/amd_gfx1250_TDM.h>)
  i32x8 gz8 = {0, 0, 0, 0, 0, 0, 0, 0};
  __builtin_amdgcn_tensor_load_to_lds(g0, g1, gz, gz, gz8, 0);
#else
  __builtin_amdgcn_tensor_load_to_lds(g0, g1, gz, gz, 0);
#endif
}
#else
#define HAVE_TDM 0
#endif

// ---------------- kernel 0: discretization tables (tiny) --------------------
__global__ void build_tables_kernel(const float* __restrict__ Are,
                                    const float* __restrict__ Aim,
                                    float* VLre2, float* VLim2m, float* Wf,
                                    float* Ere2, float* Eim2m,
                                    float* dALre, float* dALim,
                                    float* dBre, float* dBim) {
  const int n = threadIdx.x;
  if (n >= HID) return;
  const float ar = Are[n], ai = Aim[n];
  const float hr = 1.f - 0.5f * DT_ * ar, hi = -0.5f * DT_ * ai;  // 1 - DT/2 A
  const float gr = 1.f + 0.5f * DT_ * ar, gi = 0.5f * DT_ * ai;   // 1 + DT/2 A
  const float den = hr * hr + hi * hi;
  const float dar = (gr * hr + gi * hi) / den;
  const float dai = (gi * hr - gr * hi) / den;
  dBre[n] = DT_ * hr / den;
  dBim[n] = -DT_ * hi / den;

  float pr[CHUNK + 1], pi[CHUNK + 1];
  pr[0] = 1.f; pi[0] = 0.f;
#pragma unroll 1
  for (int l = 0; l < CHUNK; ++l) {
    pr[l + 1] = pr[l] * dar - pi[l] * dai;
    pi[l + 1] = pr[l] * dai + pi[l] * dar;
  }
#pragma unroll 1
  for (int l = 0; l < CHUNK; ++l) {
    VLre2[n * CHUNK + l] = 2.f * pr[l];            // B[mode][lag] for Kd_s
    VLim2m[n * CHUNK + l] = -2.f * pi[l];
    Ere2[n * CHUNK + l] = 2.f * pr[l + 1];         // B[mode][r] for y_inter
    Eim2m[n * CHUNK + l] = -2.f * pi[l + 1];
    Wf[l * 128 + n] = pr[CHUNK - 1 - l];           // B[lag][mode re|im]
    Wf[l * 128 + HID + n] = pi[CHUNK - 1 - l];
  }
  dALre[n] = pr[CHUNK];
  dALim[n] = pi[CHUNK];
}

// ---------------- kernel 1: fused QKV projection (WMMA) ---------------------
__global__ void __launch_bounds__(128)
qkv_gemm_kernel(const float* __restrict__ x, const float* __restrict__ Wq,
                const float* __restrict__ Wk, const float* __restrict__ Wv,
                float* __restrict__ Q, float* __restrict__ K,
                float* __restrict__ V) {
  const int lane = threadIdx.x;
  const int n0 = (blockIdx.x * 4 + threadIdx.y) * 16;
  const int m0 = blockIdx.y * 16;
  v8f cq = {}, ck = {}, cv = {};
#pragma unroll 4
  for (int k0 = 0; k0 < D_DIM; k0 += 32) {
    __builtin_prefetch(x + m0 * D_DIM + k0 + 128, 0, 1);  // global_prefetch_b8
    v16h a = load_a_frag(
        [&](int m, int k) { return (_Float16)x[(m0 + m) * D_DIM + k0 + k]; },
        lane);
    v16h bq = load_b_frag(
        [&](int k, int n) { return (_Float16)Wq[(n0 + n) * D_DIM + k0 + k]; },
        lane);
    v16h bk = load_b_frag(
        [&](int k, int n) { return (_Float16)Wk[(n0 + n) * D_DIM + k0 + k]; },
        lane);
    v16h bv = load_b_frag(
        [&](int k, int n) { return (_Float16)Wv[(n0 + n) * D_DIM + k0 + k]; },
        lane);
    cq = wmma_f16(a, bq, cq);
    ck = wmma_f16(a, bk, ck);
    cv = wmma_f16(a, bv, cv);
  }
  store_c_frag(cq, [&](int m, int n, float v) { Q[(m0 + m) * D_DIM + n0 + n] = v; }, lane);
  store_c_frag(ck, [&](int m, int n, float v) { K[(m0 + m) * D_DIM + n0 + n] = v; }, lane);
  store_c_frag(cv, [&](int m, int n, float v) { V[(m0 + m) * D_DIM + n0 + n] = v; }, lane);
}

// ---------------- kernel 2: shift-SSM (depthwise 64-tap causal FIR) ---------
__global__ void shift_fir_kernel(const float* __restrict__ K,
                                 const float* __restrict__ Csh,
                                 const float* __restrict__ Dsh,
                                 float* __restrict__ Ks) {
  const int i = blockIdx.x * 256 + threadIdx.x;
  const int t = i >> 9, c = i & (D_DIM - 1);
  float acc = Dsh[c] * K[t * D_DIM + c];
  const int jm = t < (HID - 1) ? t : (HID - 1);
  for (int j = 0; j <= jm; ++j)
    acc += Csh[c * HID + j] * K[(t - j) * D_DIM + c];
  Ks[i] = acc;
}

// ---------------- kernel 3: fused chunked S4D block + Q contraction ---------
// One workgroup owns (head h, 16 c-channels, 8 d-channels) = 128 (c,d) pairs,
// keeps complex SSM state H (128x64) resident in LDS across all 16 chunks.
#define NCH 128
// dynamic-LDS byte offsets (dynamic LDS starts at 0: no static __shared__)
#define OFF_HRE  0u
#define OFF_HIM  (OFF_HRE + NCH * HID * 4u)
#define OFF_CBRE (OFF_HIM + NCH * HID * 4u)
#define OFF_CBIM (OFF_CBRE + NCH * HID * 2u)
#define OFF_KDS  (OFF_CBIM + NCH * HID * 2u)
#define OFF_U    (OFF_KDS + NCH * CHUNK * 2u)
#define OFF_KSS  (OFF_U + NCH * CHUNK * 2u)
#define OFF_VS   (OFF_KSS + CHUNK * 16u * 4u)
#define OFF_QS   (OFF_VS + CHUNK * 8u * 4u)
#define OFF_OT   (OFF_QS + CHUNK * 16u * 4u)
#define OFF_DD   (OFF_OT + CHUNK * 8u * 4u)
#define SMEM_SZ  (OFF_DD + NCH * 4u)   // ~140.5 KB of 320 KB WGP LDS

__global__ void __launch_bounds__(256)
diag_fused_kernel(const float* __restrict__ Q, const float* __restrict__ Ksh,
                  const float* __restrict__ V, const float* __restrict__ Cre,
                  const float* __restrict__ Cim,
                  const float* __restrict__ Ddiag,
                  const float* __restrict__ VLre2,
                  const float* __restrict__ VLim2m,
                  const float* __restrict__ Wf, const float* __restrict__ Ere2,
                  const float* __restrict__ Eim2m,
                  const float* __restrict__ dALre,
                  const float* __restrict__ dALim,
                  const float* __restrict__ dBre,
                  const float* __restrict__ dBim, float* __restrict__ Obuf) {
  extern __shared__ char smem[];
  float* Hre = (float*)(smem + OFF_HRE);
  float* Him = (float*)(smem + OFF_HIM);
  _Float16* CBre = (_Float16*)(smem + OFF_CBRE);
  _Float16* CBim = (_Float16*)(smem + OFF_CBIM);
  _Float16* Kds = (_Float16*)(smem + OFF_KDS);
  _Float16* U = (_Float16*)(smem + OFF_U);
  float* KsS = (float*)(smem + OFF_KSS);
  float* VS = (float*)(smem + OFF_VS);
  float* QS = (float*)(smem + OFF_QS);
  float* Otile = (float*)(smem + OFF_OT);
  float* DdS = (float*)(smem + OFF_DD);

  const int tid = threadIdx.x, lane = tid & 31, wave = tid >> 5;
  const int h = blockIdx.x >> 5;
  const int ct = (blockIdx.x >> 3) & 3;
  const int dt = blockIdx.x & 7;
  const int cd_base = h * (DH * DH) + ct * 16 * DH + dt * 8;

  // ---- one-time init: H=0, CB = (Cre + i Cim) * dB, D_diag slab -----------
  for (int i = tid; i < NCH * HID; i += 256) {
    Hre[i] = 0.f; Him[i] = 0.f;
    const int ch = i >> 6, n = i & 63;
    const int cd = cd_base + (ch >> 3) * DH + (ch & 7);
    const float cr = Cre[cd * HID + n], ci = Cim[cd * HID + n];
    CBre[i] = (_Float16)(cr * dBre[n] - ci * dBim[n]);
    CBim[i] = (_Float16)(cr * dBim[n] + ci * dBre[n]);
  }
  for (int i = tid; i < NCH; i += 256)
    DdS[i] = Ddiag[cd_base + (i >> 3) * DH + (i & 7)];
  for (int i = tid; i < CHUNK * 8; i += 256) Otile[i] = 0.f;
  __syncthreads();

  // ---- Kd_short (128x64) = 2*Re(CB @ dA^l) via WMMA -----------------------
  for (int job = wave; job < 32; job += 8) {
    const int m0 = (job >> 2) * 16, n0 = (job & 3) * 16;
    v8f acc = {};
#pragma unroll
    for (int ks = 0; ks < HID; ks += 32) {
      v16h ar = load_a_frag(
          [&](int m, int k) { return CBre[(m0 + m) * HID + ks + k]; }, lane);
      v16h br = load_b_frag(
          [&](int k, int n) { return (_Float16)VLre2[(ks + k) * CHUNK + n0 + n]; },
          lane);
      acc = wmma_f16(ar, br, acc);
      v16h ai = load_a_frag(
          [&](int m, int k) { return CBim[(m0 + m) * HID + ks + k]; }, lane);
      v16h bi = load_b_frag(
          [&](int k, int n) { return (_Float16)VLim2m[(ks + k) * CHUNK + n0 + n]; },
          lane);
      acc = wmma_f16(ai, bi, acc);
    }
    store_c_frag(acc,
                 [&](int m, int n, float v) {
                   Kds[(m0 + m) * CHUNK + n0 + n] = (_Float16)v;
                 },
                 lane);
  }
  __syncthreads();

  // ---- sequential chunk loop (state H stays in LDS) -----------------------
  for (int j = 0; j < NCHUNK; ++j) {
    const int t0 = j * CHUNK;
#if HAVE_TDM
    // TDM: three strided 2-D tiles global->LDS, issued once by wave 0.
    if (wave == 0) {
      tdm_load_tile_2d(OFF_KSS, Ksh + (size_t)t0 * D_DIM + h * DH + ct * 16,
                       16, CHUNK, D_DIM);
      tdm_load_tile_2d(OFF_QS, Q + (size_t)t0 * D_DIM + h * DH + ct * 16,
                       16, CHUNK, D_DIM);
      tdm_load_tile_2d(OFF_VS, V + (size_t)t0 * D_DIM + h * DH + dt * 8,
                       8, CHUNK, D_DIM);
      __builtin_amdgcn_s_wait_tensorcnt(0);
    }
#else
    for (int i = tid; i < CHUNK * 16; i += 256) {
      const int r = i >> 4, c = i & 15;
      const int col = h * DH + ct * 16 + c;
      KsS[i] = Ksh[(t0 + r) * D_DIM + col];
      QS[i] = Q[(t0 + r) * D_DIM + col];
    }
    for (int i = tid; i < CHUNK * 8; i += 256) {
      const int r = i >> 3, d = i & 7;
      VS[i] = V[(t0 + r) * D_DIM + h * DH + dt * 8 + d];
    }
#endif
    __syncthreads();
    // regenerate U[ch, l] = K[t,c] * V[t,d] (never materialized in HBM)
    for (int i = tid; i < NCH * CHUNK; i += 256) {
      const int ch = i >> 6, l = i & 63;
      U[i] = (_Float16)(KsS[l * 16 + (ch >> 3)] * VS[l * 8 + (ch & 7)]);
    }
    __syncthreads();

    // ---- y = y_inter(WMMA on H) + y_intra(FIR) + skip, then Q-contract ----
    for (int job = wave; job < 32; job += 8) {
      const int m0 = (job >> 2) * 16, n0 = (job & 3) * 16;
      v8f acc = {};
#pragma unroll
      for (int ks = 0; ks < HID; ks += 32) {
        v16h apr = load_a_frag(
            [&](int m, int k) {
              const int idx = (m0 + m) * HID + ks + k;
              return (_Float16)((float)CBre[idx] * Hre[idx] -
                                (float)CBim[idx] * Him[idx]);
            },
            lane);
        v16h ber = load_b_frag(
            [&](int k, int n) { return (_Float16)Ere2[(ks + k) * CHUNK + n0 + n]; },
            lane);
        acc = wmma_f16(apr, ber, acc);
        v16h api = load_a_frag(
            [&](int m, int k) {
              const int idx = (m0 + m) * HID + ks + k;
              return (_Float16)((float)CBre[idx] * Him[idx] +
                                (float)CBim[idx] * Hre[idx]);
            },
            lane);
        v16h bei = load_b_frag(
            [&](int k, int n) { return (_Float16)Eim2m[(ks + k) * CHUNK + n0 + n]; },
            lane);
        acc = wmma_f16(api, bei, acc);
      }
      const int half = lane >> 4, nc = lane & 15;
#pragma unroll
      for (int e = 0; e < 8; ++e) {
        const int ch = m0 + e + 8 * half;  // (c,d) channel in slab
        const int r = n0 + nc;             // position in chunk
        float y = acc[e];
        for (int lag = 0; lag <= r; ++lag)
          y += (float)Kds[ch * CHUNK + lag] * (float)U[ch * CHUNK + (r - lag)];
        y += DdS[ch] * (float)U[ch * CHUNK + r];  // D_diag skip
        const float qv = QS[r * 16 + (ch >> 3)];
        atomicAdd(&Otile[r * 8 + (ch & 7)], qv * y);  // ds_add_f32 c-reduce
      }
    }
    __syncthreads();

    // ---- state update: H <- dA^L * H + U @ dA^{63-l} (WMMA, re/im paired) -
    for (int job = wave; job < 32; job += 8) {
      const int m0 = (job >> 2) * 16, n0 = (job & 3) * 16;
      v8f accr = {}, acci = {};
#pragma unroll
      for (int ks = 0; ks < CHUNK; ks += 32) {
        v16h au = load_a_frag(
            [&](int m, int k) { return U[(m0 + m) * CHUNK + ks + k]; }, lane);
        v16h bwr = load_b_frag(
            [&](int k, int n) { return (_Float16)Wf[(ks + k) * 128 + n0 + n]; },
            lane);
        accr = wmma_f16(au, bwr, accr);
        v16h bwi = load_b_frag(
            [&](int k, int n) {
              return (_Float16)Wf[(ks + k) * 128 + HID + n0 + n];
            },
            lane);
        acci = wmma_f16(au, bwi, acci);
      }
      const int half = lane >> 4, nc = lane & 15;
#pragma unroll
      for (int e = 0; e < 8; ++e) {
        const int ch = m0 + e + 8 * half;
        const int mode = n0 + nc;
        const int idx = ch * HID + mode;
        const float hr = Hre[idx], hi = Him[idx];
        const float ar = dALre[mode], ai = dALim[mode];
        Hre[idx] = ar * hr - ai * hi + accr[e];
        Him[idx] = ar * hi + ai * hr + acci[e];
      }
    }
    __syncthreads();

    // ---- flush O tile for this chunk (cross-c-tile reduction in HBM) ------
    for (int i = tid; i < CHUNK * 8; i += 256) {
      const int r = i >> 3, d = i & 7;
      atomicAdd(&Obuf[(t0 + r) * D_DIM + h * DH + dt * 8 + d], Otile[i]);
      Otile[i] = 0.f;
    }
    __syncthreads();
  }
}

// ---------------- kernel 4: output projection (WMMA) ------------------------
__global__ void __launch_bounds__(128)
out_gemm_kernel(const float* __restrict__ O, const float* __restrict__ Wo,
                float* __restrict__ out) {
  const int lane = threadIdx.x;
  const int n0 = (blockIdx.x * 4 + threadIdx.y) * 16;
  const int m0 = blockIdx.y * 16;
  v8f c = {};
#pragma unroll 4
  for (int k0 = 0; k0 < D_DIM; k0 += 32) {
    __builtin_prefetch(O + m0 * D_DIM + k0 + 128, 0, 1);
    v16h a = load_a_frag(
        [&](int m, int k) { return (_Float16)O[(m0 + m) * D_DIM + k0 + k]; },
        lane);
    v16h b = load_b_frag(
        [&](int k, int n) { return (_Float16)Wo[(n0 + n) * D_DIM + k0 + k]; },
        lane);
    c = wmma_f16(a, b, c);
  }
  store_c_frag(c, [&](int m, int n, float v) { out[(m0 + m) * D_DIM + n0 + n] = v; },
               lane);
}

// ---------------- host launcher --------------------------------------------
extern "C" void kernel_launch(void* const* d_in, const int* in_sizes, int n_in,
                              void* d_out, int out_size, void* d_ws,
                              size_t ws_size, hipStream_t stream) {
  const float* x = (const float*)d_in[0];
  const float* WQ = (const float*)d_in[1];
  const float* WK = (const float*)d_in[2];
  const float* WV = (const float*)d_in[3];
  const float* WO = (const float*)d_in[4];
  const float* Csh = (const float*)d_in[5];
  const float* Dsh = (const float*)d_in[6];
  const float* Are = (const float*)d_in[7];
  const float* Aim = (const float*)d_in[8];
  const float* Cre = (const float*)d_in[9];
  const float* Cim = (const float*)d_in[10];
  const float* Ddg = (const float*)d_in[11];
  float* out = (float*)d_out;

  float* ws = (float*)d_ws;
  const size_t SD = (size_t)S_LEN * D_DIM;  // 524288
  float* Qbuf = ws;
  float* Kbuf = Qbuf + SD;
  float* Ksbuf = Kbuf + SD;
  float* Vbuf = Ksbuf + SD;
  float* Obuf = Vbuf + SD;
  float* VLre2 = Obuf + SD;
  float* VLim2m = VLre2 + HID * CHUNK;
  float* Wf = VLim2m + HID * CHUNK;
  float* Ere2 = Wf + CHUNK * 128;
  float* Eim2m = Ere2 + HID * CHUNK;
  float* dALre = Eim2m + HID * CHUNK;
  float* dALim = dALre + HID;
  float* dBre = dALim + HID;
  float* dBim = dBre + HID;

  hipMemsetAsync(Obuf, 0, SD * sizeof(float), stream);

  build_tables_kernel<<<1, 64, 0, stream>>>(Are, Aim, VLre2, VLim2m, Wf, Ere2,
                                            Eim2m, dALre, dALim, dBre, dBim);

  qkv_gemm_kernel<<<dim3(D_DIM / 64, S_LEN / 16), dim3(32, 4), 0, stream>>>(
      x, WQ, WK, WV, Qbuf, Kbuf, Vbuf);

  shift_fir_kernel<<<(S_LEN * D_DIM) / 256, 256, 0, stream>>>(Kbuf, Csh, Dsh,
                                                              Ksbuf);

  diag_fused_kernel<<<HEADS * 4 * 8, 256, SMEM_SZ, stream>>>(
      Qbuf, Ksbuf, Vbuf, Cre, Cim, Ddg, VLre2, VLim2m, Wf, Ere2, Eim2m, dALre,
      dALim, dBre, dBim, Obuf);

  out_gemm_kernel<<<dim3(D_DIM / 64, S_LEN / 16), dim3(32, 4), 0, stream>>>(
      Obuf, WO, out);
}
